// BambaMixerDecoderLayer_84696755077458
// MI455X (gfx1250) — matrix operand order repaired and checked by reading, not compiled, not run on hardware
//
#include <hip/hip_runtime.h>

// ---------------------------------------------------------------- types
typedef unsigned short u16;
typedef unsigned int   u32;
typedef __attribute__((ext_vector_type(16))) __bf16 v16bf;
typedef __attribute__((ext_vector_type(8)))  float  v8f;
typedef __attribute__((ext_vector_type(4)))  u32    u32x4;

union Frag {          // 16 bf16 = 32 bytes = two 128-bit LDS loads
    v16bf v;
    u32x4 q[2];
};

// ---------------------------------------------------------------- sizes
#define HDIM     2048
#define DIN      4096
#define DSTATE   128
#define DCONVW   4
#define NHEAD    64
#define HEADD    64
#define FFDIM    8192
#define CONVD    (DIN + 2*DSTATE)            // 4352
#define DPROJ    (2*DIN + 2*DSTATE + NHEAD)  // 8512
#define BATCH    2
#define SEQ      2048
#define MTOK     (BATCH*SEQ)                 // 4096
#define EPSI     1e-5f

#define BM 128
#define BN 64
#define BK 32

// ---------------------------------------------------------------- helpers
__device__ __forceinline__ u16 f2bf(float f) {
    u32 u = __float_as_uint(f);
    u += 0x7FFFu + ((u >> 16) & 1u);   // round-to-nearest-even
    return (u16)(u >> 16);
}
__device__ __forceinline__ float silu_f(float x) {
    return x / (1.0f + __expf(-x));
}
__device__ __forceinline__ u32 lds_off(const void* p) {
    // generic LDS pointer: low 32 bits are the LDS byte offset (ISA 10.2 aperture map)
    return (u32)(unsigned long long)p;
}
// CDNA5 async DMA: global -> LDS, tracked with ASYNCcnt (ISA 15.18.3 / 08_async_tensor)
__device__ __forceinline__ void async_cp_b128(u32 lds, const void* g) {
    asm volatile("global_load_async_to_lds_b128 %0, %1, off"
                 :: "v"(lds), "v"((unsigned long long)g) : "memory");
}
__device__ __forceinline__ void async_cp_b32(u32 lds, const void* g) {
    asm volatile("global_load_async_to_lds_b32 %0, %1, off"
                 :: "v"(lds), "v"((unsigned long long)g) : "memory");
}
__device__ __forceinline__ void wait_async0() { asm volatile("s_wait_asynccnt 0x0" ::: "memory"); }
__device__ __forceinline__ void wait_async3() { asm volatile("s_wait_asynccnt 0x3" ::: "memory"); }
__device__ __forceinline__ void wait_async4() { asm volatile("s_wait_asynccnt 0x4" ::: "memory"); }

__device__ __forceinline__ float block_sum(float v, float* red) {
    #pragma unroll
    for (int o = 16; o > 0; o >>= 1) v += __shfl_xor(v, o, 32);
    int w = threadIdx.x >> 5;
    if ((threadIdx.x & 31) == 0) red[w] = v;
    __syncthreads();
    if (threadIdx.x == 0) {
        float s = 0.f;
        for (int i = 0; i < (int)(blockDim.x >> 5); ++i) s += red[i];
        red[0] = s;
    }
    __syncthreads();
    return red[0];
}

// ---------------------------------------------------------------- cast + transpose
// in: K x N f32 row-major  ->  out: N x K bf16 row-major (so GEMM B tiles are
// K-contiguous per column and can be staged with straight b128 async copies)
__global__ __launch_bounds__(256) void cast_transpose_kernel(const float* __restrict__ in,
                                                             u16* __restrict__ out,
                                                             int K, int N) {
    __shared__ u16 tile[32][33];
    const int k0 = blockIdx.x * 32, n0 = blockIdx.y * 32;
    const int tn = threadIdx.x & 31, ti = threadIdx.x >> 5;   // 8 rows per pass
    #pragma unroll
    for (int r = 0; r < 4; ++r) {
        int k = ti + r * 8;
        tile[k][tn] = f2bf(in[(long)(k0 + k) * N + n0 + tn]);
    }
    __syncthreads();
    #pragma unroll
    for (int r = 0; r < 4; ++r) {
        int n = ti + r * 8;
        out[(long)(n0 + n) * K + k0 + tn] = tile[tn][n];
    }
}

// ---------------------------------------------------------------- rmsnorm -> bf16
__global__ __launch_bounds__(256) void rmsnorm_bf16_kernel(const float* __restrict__ x,
                                                           const float* __restrict__ w,
                                                           u16* __restrict__ out, int D) {
    __shared__ float red[8];
    long m = blockIdx.x;
    const float* row = x + m * (long)D;
    float part = 0.f;
    for (int d = threadIdx.x; d < D; d += 256) { float v = row[d]; part += v * v; }
    float tot = block_sum(part, red);
    float r = rsqrtf(tot / (float)D + EPSI);
    for (int d = threadIdx.x; d < D; d += 256)
        out[m * (long)D + d] = f2bf(row[d] * r * w[d]);
}

// ---------------------------------------------------------------- WMMA GEMM
// A : M x K bf16 row-major.  Bt : N x K bf16 row-major (pre-transposed weights).
// C : M x N f32.   MODE 0: C = A*Bt^T    MODE 1: C = A*Bt^T + resid
// Double-buffered LDS staged entirely with global_load_async_to_lds_b128.
template<int MODE>
__global__ __launch_bounds__(256)
void gemm_bf16_wmma(const u16* __restrict__ A, const u16* __restrict__ Bt,
                    float* __restrict__ C, const float* __restrict__ resid,
                    int K, int N) {
    __shared__ __align__(16) u16 sA[2][BM * BK];   // [m][k]
    __shared__ __align__(16) u16 sB[2][BN * BK];   // [n][k]
    const int t    = threadIdx.x;
    const int wid  = t >> 5, lane = t & 31;
    const int half = lane >> 4, l16 = lane & 15;
    const int bm = blockIdx.y * BM, bn = blockIdx.x * BN;
    const int wm = (wid >> 1) * 32, wn = (wid & 1) * 32;

    // per-thread staging chunks (16B each): A has 512 chunks, B has 256
    const int am0 = t >> 2,          ak0 = (t & 3) * 8;          // chunk t
    const int am1 = (t + 256) >> 2,  ak1 = ak0;                  // chunk t+256
    const int bnn = t >> 2,          bkk = (t & 3) * 8;

    auto stage = [&](int buf, int k0) {
        async_cp_b128(lds_off(&sA[buf][am0 * BK + ak0]), A + (long)(bm + am0) * K + k0 + ak0);
        async_cp_b128(lds_off(&sA[buf][am1 * BK + ak1]), A + (long)(bm + am1) * K + k0 + ak1);
        async_cp_b128(lds_off(&sB[buf][bnn * BK + bkk]), Bt + (long)(bn + bnn) * K + k0 + bkk);
    };

    v8f acc[2][2] = {{{}, {}}, {{}, {}}};

    stage(0, 0);
    int cur = 0;
    for (int k0 = 0; k0 < K; k0 += BK) {
        const bool has_next = (k0 + BK < K);
        if (has_next) stage(cur ^ 1, k0 + BK);
        if (has_next) wait_async3(); else wait_async0();
        __syncthreads();                       // tile `cur` complete in LDS (all waves)

        Frag a[2], b[2];
        #pragma unroll
        for (int i = 0; i < 2; ++i) {          // A: lane=m, K = half*16 + 0..15 (same order as B)
            const u32x4* p = (const u32x4*)&sA[cur][(wm + i * 16 + l16) * BK];
            a[i].q[0] = p[2 * half];           // K = half*16 + 0..7
            a[i].q[1] = p[2 * half + 1];       // K = half*16 + 8..15
        }
        #pragma unroll
        for (int j = 0; j < 2; ++j) {          // B: lane=n, K = half*16 + 0..15
            const u32x4* p = (const u32x4*)&sB[cur][(wn + j * 16 + l16) * BK];
            b[j].q[0] = p[2 * half];
            b[j].q[1] = p[2 * half + 1];
        }
        #pragma unroll
        for (int i = 0; i < 2; ++i)
            #pragma unroll
            for (int j = 0; j < 2; ++j)
                acc[i][j] = __builtin_amdgcn_wmma_f32_16x16x32_bf16(
                    false, a[i].v, false, b[j].v, (short)0, acc[i][j], false, false);
        __syncthreads();                       // done reading `cur` before it is re-staged
        cur ^= 1;
    }

    #pragma unroll
    for (int i = 0; i < 2; ++i)
        #pragma unroll
        for (int j = 0; j < 2; ++j) {
            int col = bn + wn + j * 16 + l16;
            #pragma unroll
            for (int r = 0; r < 8; ++r) {
                int row = bm + wm + i * 16 + r + half * 8;
                long off = (long)row * N + col;
                float v = acc[i][j][r];
                if (MODE == 1) v += resid[off];
                C[off] = v;
            }
        }
}

// ---------------------------------------------------------------- fused gate/up GEMM
// Bt: (2*FF) x K bf16 (transposed w_gate_up); rows [0,FF)=gate, [FF,2FF)=up.
// out = bf16( silu(gate) * up ), M x FF
__global__ __launch_bounds__(256)
void gemm_gateup_wmma(const u16* __restrict__ A, const u16* __restrict__ Bt,
                      u16* __restrict__ outb, int K) {
    __shared__ __align__(16) u16 sA[2][BM * BK];
    __shared__ __align__(16) u16 sG[2][BN * BK];
    __shared__ __align__(16) u16 sU[2][BN * BK];
    const int t    = threadIdx.x;
    const int wid  = t >> 5, lane = t & 31;
    const int half = lane >> 4, l16 = lane & 15;
    const int bm = blockIdx.y * BM, bn = blockIdx.x * BN;
    const int wm = (wid >> 1) * 32, wn = (wid & 1) * 32;

    const int am0 = t >> 2,         ak0 = (t & 3) * 8;
    const int am1 = (t + 256) >> 2, ak1 = ak0;
    const int bnn = t >> 2,         bkk = (t & 3) * 8;

    auto stage = [&](int buf, int k0) {
        async_cp_b128(lds_off(&sA[buf][am0 * BK + ak0]), A + (long)(bm + am0) * K + k0 + ak0);
        async_cp_b128(lds_off(&sA[buf][am1 * BK + ak1]), A + (long)(bm + am1) * K + k0 + ak1);
        async_cp_b128(lds_off(&sG[buf][bnn * BK + bkk]), Bt + (long)(bn + bnn) * K + k0 + bkk);
        async_cp_b128(lds_off(&sU[buf][bnn * BK + bkk]),
                      Bt + (long)(FFDIM + bn + bnn) * K + k0 + bkk);
    };

    v8f accg[2][2] = {{{}, {}}, {{}, {}}};
    v8f accu[2][2] = {{{}, {}}, {{}, {}}};

    stage(0, 0);
    int cur = 0;
    for (int k0 = 0; k0 < K; k0 += BK) {
        const bool has_next = (k0 + BK < K);
        if (has_next) stage(cur ^ 1, k0 + BK);
        if (has_next) wait_async4(); else wait_async0();
        __syncthreads();

        Frag a[2], bg[2], bu[2];
        #pragma unroll
        for (int i = 0; i < 2; ++i) {
            const u32x4* p = (const u32x4*)&sA[cur][(wm + i * 16 + l16) * BK];
            a[i].q[0] = p[2 * half];
            a[i].q[1] = p[2 * half + 1];
        }
        #pragma unroll
        for (int j = 0; j < 2; ++j) {
            const u32x4* pg = (const u32x4*)&sG[cur][(wn + j * 16 + l16) * BK];
            const u32x4* pu = (const u32x4*)&sU[cur][(wn + j * 16 + l16) * BK];
            bg[j].q[0] = pg[2 * half]; bg[j].q[1] = pg[2 * half + 1];
            bu[j].q[0] = pu[2 * half]; bu[j].q[1] = pu[2 * half + 1];
        }
        #pragma unroll
        for (int i = 0; i < 2; ++i)
            #pragma unroll
            for (int j = 0; j < 2; ++j) {
                accg[i][j] = __builtin_amdgcn_wmma_f32_16x16x32_bf16(
                    false, a[i].v, false, bg[j].v, (short)0, accg[i][j], false, false);
                accu[i][j] = __builtin_amdgcn_wmma_f32_16x16x32_bf16(
                    false, a[i].v, false, bu[j].v, (short)0, accu[i][j], false, false);
            }
        __syncthreads();
        cur ^= 1;
    }

    #pragma unroll
    for (int i = 0; i < 2; ++i)
        #pragma unroll
        for (int j = 0; j < 2; ++j) {
            int col = bn + wn + j * 16 + l16;
            #pragma unroll
            for (int r = 0; r < 8; ++r) {
                int row = bm + wm + i * 16 + r + half * 8;
                float g = accg[i][j][r];
                float u = accu[i][j][r];
                outb[(long)row * FFDIM + col] = f2bf(silu_f(g) * u);
            }
        }
}

// ---------------------------------------------------------------- depthwise causal conv + SiLU
__global__ __launch_bounds__(256) void conv_silu_kernel(const float* __restrict__ proj,
                                                        const float* __restrict__ cw,
                                                        const float* __restrict__ cb,
                                                        float* __restrict__ out) {
    long idx = (long)blockIdx.x * 256 + threadIdx.x;
    if (idx >= (long)MTOK * CONVD) return;
    int c = (int)(idx % CONVD);
    long m = idx / CONVD;
    int b = (int)(m / SEQ), l = (int)(m % SEQ);
    float acc = cb[c];
    #pragma unroll
    for (int j = 0; j < DCONVW; ++j) {
        int ls = l - (DCONVW - 1) + j;
        if (ls >= 0)
            acc += cw[c * DCONVW + j] * proj[((long)b * SEQ + ls) * DPROJ + DIN + c];
    }
    out[m * (long)CONVD + c] = silu_f(acc);
}

// ---------------------------------------------------------------- dt softplus + dA
__global__ __launch_bounds__(256) void dt_da_kernel(const float* __restrict__ proj,
                                                    const float* __restrict__ dt_bias,
                                                    const float* __restrict__ A_log,
                                                    float* __restrict__ dts,
                                                    float* __restrict__ dAb) {
    long idx = (long)blockIdx.x * 256 + threadIdx.x;
    if (idx >= (long)MTOK * NHEAD) return;
    long m = idx / NHEAD;
    int h = (int)(idx % NHEAD);
    float x = proj[m * (long)DPROJ + DIN + CONVD + h] + dt_bias[h];
    float dt = (x > 20.f) ? x : log1pf(__expf(x));
    float Ah = -__expf(A_log[h]);
    dts[idx] = dt;
    dAb[idx] = __expf(dt * Ah);
}

// ---------------------------------------------------------------- SSM scan
// one block per (batch, head); state 64x128 f32 in registers: 32 per lane;
// per-step B/C/x broadcast staged to LDS with async b32 DMA.
__global__ __launch_bounds__(256) void ssm_scan_kernel(const float* __restrict__ xBC,
                                                       const float* __restrict__ dts,
                                                       const float* __restrict__ dAb,
                                                       const float* __restrict__ D_ssm,
                                                       float* __restrict__ y) {
    const int blk = blockIdx.x;
    const int b = blk / NHEAD, hh = blk % NHEAD;
    const int t = threadIdx.x;
    const int p = t >> 2;               // 0..63  (state row)
    const int sq = (t & 3) * 32;        // state col chunk
    __shared__ float sB[DSTATE], sC[DSTATE], sX[HEADD];

    float S[32];
    #pragma unroll
    for (int i = 0; i < 32; ++i) S[i] = 0.f;
    const float Dh = D_ssm[hh];

    for (int l = 0; l < SEQ; ++l) {
        const long m = (long)b * SEQ + l;
        const float* row = xBC + m * CONVD;
        if (t < DSTATE) async_cp_b32(lds_off(&sB[t]), row + DIN + t);
        else            async_cp_b32(lds_off(&sC[t - DSTATE]), row + DIN + DSTATE + (t - DSTATE));
        if (t < HEADD)  async_cp_b32(lds_off(&sX[t]), row + hh * HEADD + t);
        wait_async0();
        __syncthreads();

        const float dAt = dAb[m * NHEAD + hh];
        const float dtt = dts[m * NHEAD + hh];
        const float xp = sX[p];
        const float coef = dtt * xp;
        float acc = 0.f;
        #pragma unroll
        for (int i = 0; i < 32; ++i) {
            float s = S[i] * dAt + coef * sB[sq + i];
            S[i] = s;
            acc += s * sC[sq + i];
        }
        acc += __shfl_down(acc, 1, 4);
        acc += __shfl_down(acc, 2, 4);
        if ((t & 3) == 0)
            y[m * DIN + hh * HEADD + p] = acc + Dh * xp;
        __syncthreads();
    }
}

// ---------------------------------------------------------------- gated rmsnorm -> bf16
__global__ __launch_bounds__(256) void gate_norm_kernel(const float* __restrict__ proj,
                                                        const float* __restrict__ y,
                                                        const float* __restrict__ norm_w,
                                                        u16* __restrict__ out) {
    __shared__ float g[DIN];
    __shared__ float red[8];
    long m = blockIdx.x;
    int t = threadIdx.x;
    const float* zrow = proj + m * (long)DPROJ;
    const float* yrow = y + m * (long)DIN;
    float part = 0.f;
    #pragma unroll
    for (int i = 0; i < DIN / 256; ++i) {
        int d = t + i * 256;
        float z = zrow[d];
        float gg = yrow[d] * silu_f(z);
        g[d] = gg;
        part += gg * gg;
    }
    float tot = block_sum(part, red);
    float r = rsqrtf(tot / (float)DIN + EPSI);
    #pragma unroll
    for (int i = 0; i < DIN / 256; ++i) {
        int d = t + i * 256;
        out[m * (long)DIN + d] = f2bf(g[d] * r * norm_w[d]);
    }
}

// ---------------------------------------------------------------- launch
extern "C" void kernel_launch(void* const* d_in, const int* in_sizes, int n_in,
                              void* d_out, int out_size, void* d_ws, size_t ws_size,
                              hipStream_t stream) {
    const float* hidden   = (const float*)d_in[0];
    const float* w_in     = (const float*)d_in[1];
    const float* conv_w   = (const float*)d_in[2];
    const float* conv_b   = (const float*)d_in[3];
    const float* A_log    = (const float*)d_in[4];
    const float* D_ssm    = (const float*)d_in[5];
    const float* dt_bias  = (const float*)d_in[6];
    const float* norm_w   = (const float*)d_in[7];
    const float* w_out    = (const float*)d_in[8];
    const float* ln1_w    = (const float*)d_in[9];
    const float* ln2_w    = (const float*)d_in[10];
    const float* w_gu     = (const float*)d_in[11];
    const float* w_down   = (const float*)d_in[12];

    float* out_h   = (float*)d_out;
    float* out_res = (float*)d_out + (long)MTOK * HDIM;

    char* ws = (char*)d_ws;
    size_t off = 0;
    auto alloc = [&](size_t bytes) -> void* {
        void* p = ws + off;
        off += (bytes + 255) & ~(size_t)255;
        return p;
    };
    u16*   hA     = (u16*)alloc((size_t)MTOK * HDIM * 2);
    u16*   winT   = (u16*)alloc((size_t)HDIM * DPROJ * 2);        // [DPROJ][HDIM]
    u16*   woutT  = (u16*)alloc((size_t)DIN * HDIM * 2);          // [HDIM][DIN]
    u16*   wguT   = (u16*)alloc((size_t)HDIM * 2 * FFDIM * 2);    // [2FF][HDIM]
    u16*   wdnT   = (u16*)alloc((size_t)FFDIM * HDIM * 2);        // [HDIM][FFDIM]
    float* proj   = (float*)alloc((size_t)MTOK * DPROJ * 4);
    float* xbc    = (float*)alloc((size_t)MTOK * CONVD * 4);
    float* dts    = (float*)alloc((size_t)MTOK * NHEAD * 4);
    float* dAb    = (float*)alloc((size_t)MTOK * NHEAD * 4);
    float* ybuf   = (float*)alloc((size_t)MTOK * DIN * 4);
    u16*   yg_b   = (u16*)alloc((size_t)MTOK * DIN * 2);
    u16*   h2_b   = (u16*)alloc((size_t)MTOK * HDIM * 2);
    u16*   act_b  = (u16*)alloc((size_t)MTOK * FFDIM * 2);

    // 1) weights -> bf16, transposed to [N][K]
    cast_transpose_kernel<<<dim3(HDIM / 32, DPROJ / 32), 256, 0, stream>>>(w_in, winT, HDIM, DPROJ);
    cast_transpose_kernel<<<dim3(DIN / 32, HDIM / 32), 256, 0, stream>>>(w_out, woutT, DIN, HDIM);
    cast_transpose_kernel<<<dim3(HDIM / 32, (2 * FFDIM) / 32), 256, 0, stream>>>(w_gu, wguT, HDIM, 2 * FFDIM);
    cast_transpose_kernel<<<dim3(FFDIM / 32, HDIM / 32), 256, 0, stream>>>(w_down, wdnT, FFDIM, HDIM);

    // 2) ln1 rmsnorm -> bf16
    rmsnorm_bf16_kernel<<<MTOK, 256, 0, stream>>>(hidden, ln1_w, hA, HDIM);

    // 3) in_proj: (MTOK x HDIM) x (HDIM x DPROJ) -> proj f32
    gemm_bf16_wmma<0><<<dim3(DPROJ / BN, MTOK / BM), 256, 0, stream>>>(
        hA, winT, proj, nullptr, HDIM, DPROJ);

    // 4) causal depthwise conv + SiLU
    {
        long n = (long)MTOK * CONVD;
        conv_silu_kernel<<<(int)((n + 255) / 256), 256, 0, stream>>>(proj, conv_w, conv_b, xbc);
    }
    // 5) dt softplus + dA
    {
        long n = (long)MTOK * NHEAD;
        dt_da_kernel<<<(int)((n + 255) / 256), 256, 0, stream>>>(proj, dt_bias, A_log, dts, dAb);
    }
    // 6) SSM scan
    ssm_scan_kernel<<<BATCH * NHEAD, 256, 0, stream>>>(xbc, dts, dAb, D_ssm, ybuf);

    // 7) gated rmsnorm -> bf16
    gate_norm_kernel<<<MTOK, 256, 0, stream>>>(proj, ybuf, norm_w, yg_b);

    // 8) out_proj + residual -> out_res
    gemm_bf16_wmma<1><<<dim3(HDIM / BN, MTOK / BM), 256, 0, stream>>>(
        yg_b, woutT, out_res, hidden, DIN, HDIM);

    // 9) ln2 rmsnorm -> bf16
    rmsnorm_bf16_kernel<<<MTOK, 256, 0, stream>>>(out_res, ln2_w, h2_b, HDIM);

    // 10) fused gate/up GEMM + SiLU*up -> bf16 activations
    gemm_gateup_wmma<<<dim3(FFDIM / BN, MTOK / BM), 256, 0, stream>>>(h2_b, wguT, act_b, HDIM);

    // 11) down proj -> out_h
    gemm_bf16_wmma<0><<<dim3(HDIM / BN, MTOK / BM), 256, 0, stream>>>(
        act_b, wdnT, out_h, nullptr, FFDIM, HDIM);
}
